// DenseSum_59889023976041
// MI455X (gfx1250) — matrix operand order repaired and checked by reading, not compiled, not run on hardware
//
#include <hip/hip_runtime.h>
#include <hip/hip_fp16.h>
#include <stdint.h>

// Problem sizes (from reference): S=32, D=8, B=128, N_IN=512, N_SUMS=512
#define S_DIM   32
#define D_DIM   8
#define B_DIM   128
#define NN_IN   512
#define N_SUMS  512
#define KT      128   // output-column tile per workgroup
#define NT      32    // reduction chunk = WMMA K for f16
#define AS_STRIDE 40  // padded LDS row stride (halfs), 80B = 16B-aligned rows
#define BS_STRIDE 40

typedef __attribute__((ext_vector_type(16))) _Float16 v16h;
typedef __attribute__((ext_vector_type(8)))  float    v8f;
typedef __attribute__((ext_vector_type(4)))  unsigned int v4u;
typedef __attribute__((ext_vector_type(8)))  int v8i;
typedef __attribute__((ext_vector_type(4)))  int v4i;

union Frag {
  v16h   v;
  float4 q[2];
};

#if defined(__has_builtin)
#if __has_builtin(__builtin_amdgcn_tensor_load_to_lds) && __has_builtin(__builtin_amdgcn_s_wait_tensorcnt)
#define HAVE_TDM 1
#endif
#endif
#ifndef HAVE_TDM
#define HAVE_TDM 0
#endif

#if HAVE_TDM
// Issue one TDM descriptor: load a [tile_h rows x tile_w cols] f32 tile with
// row stride `row_stride` (all in elements) into contiguous LDS at lds_off.
// Descriptor packing per CDNA5 ISA 8.3/8.4 (D# group0 = 4 dwords, group1 = 8 dwords).
// This toolchain exposes the 6-arg builtin: (u32x4, i32x8, i32x4, i32x4, i32x8, i32 cpol);
// zeroed groups 2/3 lower to the 2-SGPR-group (2-D tensor) encoding.
__device__ __forceinline__ void tdm_load_tile_f32(const float* gsrc, unsigned lds_off,
                                                  unsigned tensor_w, unsigned tensor_h,
                                                  unsigned row_stride,
                                                  unsigned tile_w, unsigned tile_h) {
  unsigned long long ga = (unsigned long long)(uintptr_t)gsrc;
  v4u g0;
  g0[0] = 1u;                                    // count=1 (valid user descriptor)
  g0[1] = lds_off;                               // lds_addr (bytes)
  g0[2] = (unsigned)ga;                          // global_addr[31:0]
  g0[3] = ((unsigned)(ga >> 32) & 0x01FFFFFFu)   // global_addr[56:32]
          | (2u << 30);                          // type = 2 ("image")
  v8i g1;
  g1[0] = (int)(2u << 16);                       // workgroup_mask=0, data_size=2 (4B)
  g1[1] = (int)(tensor_w << 16);                 // tensor_dim0[15:0] -> bits[31:16]
  g1[2] = (int)((tensor_w >> 16) | (tensor_h << 16)); // dim0 hi | tensor_dim1 lo
  g1[3] = (int)((tensor_h >> 16) | (tile_w << 16));   // dim1 hi | tile_dim0
  g1[4] = (int)tile_h;                           // tile_dim1, tile_dim2=0
  g1[5] = (int)row_stride;                       // tensor_dim0_stride[31:0]
  g1[6] = (int)((row_stride & 0xFFFFu) << 16);   // stride0[47:32]=0 | stride1[15:0]
  g1[7] = (int)(row_stride >> 16);               // tensor_dim1_stride[47:16]
  v4i gz4 = {0, 0, 0, 0};
  v8i gz8 = {0, 0, 0, 0, 0, 0, 0, 0};
  __builtin_amdgcn_tensor_load_to_lds(g0, g1, gz4, gz4, gz8, 0);
}
#endif

__global__ __launch_bounds__(256)
void logspace_dense_sum_kernel(const float* __restrict__ x,
                               const float* __restrict__ acc,
                               float* __restrict__ out) {
  __shared__ _Float16 As[B_DIM][AS_STRIDE];   // exp(x - rowmax) chunk, f16
  __shared__ _Float16 Bs[KT][BS_STRIDE];      // exp(acc - colmax), transposed, f16
  __shared__ float Wraw[NT * KT];             // raw f32 acc chunk   (TDM target)
  __shared__ float Xraw[B_DIM * NT];          // raw f32 x chunk     (TDM target)
  __shared__ float rowMax[B_DIM];
  __shared__ float colMax[KT];
  __shared__ float colNLS[KT];                // -log(colsum) == b_max
  __shared__ float pm[KT];
  __shared__ float ps[KT];

  const int tid  = threadIdx.x;
  const int lane = tid & 31;
  const int wid  = tid >> 5;

  const int kt = blockIdx.x;   // 0..3 column tile
  const int d  = blockIdx.y;
  const int s  = blockIdx.z;
  const int k0 = kt * KT;

  const float* Xsd = x   + (size_t)(s * D_DIM + d) * B_DIM * NN_IN;
  const float* Wsd = acc + (size_t)(s * D_DIM + d) * NN_IN * N_SUMS;
  float*       Osd = out + (size_t)(s * D_DIM + d) * B_DIM * N_SUMS;

  // ---------------- Phase 0: rowMax[b] = max_n x[b,n] -------------------
  {
    for (int r = 0; r < 16; ++r) {
      const int row = wid * 16 + r;
      const float* xr = Xsd + (size_t)row * NN_IN;
      float m = -INFINITY;
      #pragma unroll
      for (int j = 0; j < NN_IN / 32; ++j)
        m = fmaxf(m, xr[lane + j * 32]);
      #pragma unroll
      for (int off = 16; off >= 1; off >>= 1)
        m = fmaxf(m, __shfl_xor(m, off, 32));
      if (lane == 0) rowMax[row] = m;
    }
  }

  // ------- Phase 1: online logsumexp over n for each column k ----------
  {
    const int kc = tid & (KT - 1);
    const int hh = tid >> 7;       // two threads per column (even/odd n)
    float m = -INFINITY, ssum = 0.0f;
    for (int n = hh; n < NN_IN; n += 2) {
      float v  = Wsd[(size_t)n * N_SUMS + k0 + kc];
      float nm = fmaxf(m, v);
      ssum = ssum * __expf(m - nm) + __expf(v - nm);
      m = nm;
    }
    if (hh == 1) { pm[kc] = m; ps[kc] = ssum; }
    __syncthreads();
    if (hh == 0) {
      float m2 = pm[kc], s2 = ps[kc];
      float mm = fmaxf(m, m2);
      float sc = ssum * __expf(m - mm) + s2 * __expf(m2 - mm);
      colMax[kc] = mm;
      colNLS[kc] = -__logf(sc);    // == b_max[k]
    }
  }
  __syncthreads();

  // ---------------- Phase 2: tiled f16 WMMA GEMM -----------------------
  // wave grid 4x2: wave handles a 32-row x 64-col output tile (2x4 WMMA tiles)
  const int wm = wid & 3;
  const int wn = wid >> 2;
  const int lr = lane & 15;
  const int lh = lane >> 4;

  v8f accv[2][4];
  const v8f vzero = {0.f, 0.f, 0.f, 0.f, 0.f, 0.f, 0.f, 0.f};
  #pragma unroll
  for (int i = 0; i < 2; ++i)
    #pragma unroll
    for (int j = 0; j < 4; ++j) accv[i][j] = vzero;

  for (int nc = 0; nc < NN_IN / NT; ++nc) {
    const int n0 = nc * NT;
    __syncthreads();   // previous iteration's LDS fully consumed

#if HAVE_TDM
    if (wid == 0) {
      // Two back-to-back TDM descriptors (in-order within the wave):
      //   acc chunk [NT x KT]  f32, row stride N_SUMS -> Wraw (contiguous)
      tdm_load_tile_f32(Wsd + (size_t)n0 * N_SUMS + k0, (unsigned)(uintptr_t)&Wraw[0],
                        N_SUMS, NN_IN, N_SUMS, KT, NT);
      //   x   chunk [B  x NT]  f32, row stride NN_IN  -> Xraw (contiguous)
      tdm_load_tile_f32(Xsd + n0, (unsigned)(uintptr_t)&Xraw[0],
                        NN_IN, B_DIM, NN_IN, NT, B_DIM);
    }
    if (nc + 1 < NN_IN / NT)   // hint next x chunk (global_prefetch_b8)
      __builtin_prefetch(Xsd + (size_t)(tid & 127) * NN_IN + n0 + NT, 0, 0);
    if (wid == 0) __builtin_amdgcn_s_wait_tensorcnt(0);   // covers both loads
#else
    #pragma unroll
    for (int j = 0; j < (NT * KT) / 256; ++j) {
      int idx = tid + j * 256;
      Wraw[idx] = Wsd[(size_t)(n0 + (idx >> 7)) * N_SUMS + k0 + (idx & 127)];
    }
    #pragma unroll
    for (int j = 0; j < (B_DIM * NT) / 256; ++j) {
      int idx = tid + j * 256;
      Xraw[idx] = Xsd[(size_t)(idx >> 5) * NN_IN + n0 + (idx & 31)];
    }
#endif
    __syncthreads();   // Wraw/Xraw visible to all waves

    // A chunk: As[r][c] = f16(exp(x[r][n0+c] - rowMax[r]))   (LDS->LDS transform)
    #pragma unroll
    for (int j = 0; j < (B_DIM * NT) / 256; ++j) {
      int idx = tid + j * 256;
      int r = idx >> 5, c = idx & 31;
      As[r][c] = (_Float16)__expf(Xraw[idx] - rowMax[r]);
    }
    // B chunk, exp + transpose: Bs[c][n] = f16(exp(acc[n0+n][k0+c] - colMax[c]))
    #pragma unroll
    for (int j = 0; j < (NT * KT) / 256; ++j) {
      int idx = tid + j * 256;
      int n = idx >> 7, c = idx & 127;
      Bs[c][n] = (_Float16)__expf(Wraw[idx] - colMax[c]);
    }
    __syncthreads();

    // Fragment loads per CDNA5 16-bit WMMA VGPR layouts
    Frag aF[2], bF[4];
    #pragma unroll
    for (int mt = 0; mt < 2; ++mt) {
      // A 16x32: lanes 0-15 K0..7 / K16..23, lanes 16-31 K8..15 / K24..31
      int m = wm * 32 + mt * 16 + lr;
      aF[mt].q[0] = *(const float4*)&As[m][lh * 8];
      aF[mt].q[1] = *(const float4*)&As[m][16 + lh * 8];
    }
    #pragma unroll
    for (int nt = 0; nt < 4; ++nt) {
      // B 32x16: lanes 0-15 K0..7 / K16..23, lanes 16-31 K8..15 / K24..31
      int n = wn * 64 + nt * 16 + lr;
      bF[nt].q[0] = *(const float4*)&Bs[n][lh * 8];
      bF[nt].q[1] = *(const float4*)&Bs[n][16 + lh * 8];
    }
    #pragma unroll
    for (int mt = 0; mt < 2; ++mt)
      #pragma unroll
      for (int nt = 0; nt < 4; ++nt)
        accv[mt][nt] = __builtin_amdgcn_wmma_f32_16x16x32_f16(
            false, aF[mt].v, false, bF[nt].v,
            (short)0, accv[mt][nt], false, false);
  }

  // ---------------- Epilogue: a_max + b_max + log(prod) ----------------
  #pragma unroll
  for (int mt = 0; mt < 2; ++mt) {
    #pragma unroll
    for (int nt = 0; nt < 4; ++nt) {
      #pragma unroll
      for (int v = 0; v < 8; ++v) {
        int row = wm * 32 + mt * 16 + lh * 8 + v;   // C/D layout: M = v + (lane>=16)*8
        int col = wn * 64 + nt * 16 + lr;
        float p = accv[mt][nt][v];
        Osd[(size_t)row * N_SUMS + k0 + col] =
            rowMax[row] + colNLS[col] + __logf(p);
      }
    }
  }
}

extern "C" void kernel_launch(void* const* d_in, const int* in_sizes, int n_in,
                              void* d_out, int out_size, void* d_ws, size_t ws_size,
                              hipStream_t stream) {
  (void)in_sizes; (void)n_in; (void)out_size; (void)d_ws; (void)ws_size;
  const float* x   = (const float*)d_in[0];
  const float* acc = (const float*)d_in[1];
  float* out = (float*)d_out;
  dim3 grid(N_SUMS / KT, D_DIM, S_DIM);   // 4 x 8 x 32 = 1024 workgroups
  dim3 block(256);                        // 8 wave32s
  hipLaunchKernelGGL(logspace_dense_sum_kernel, grid, block, 0, stream,
                     x, acc, out);
}
